// WinEncoderTransformer_64544768524602
// MI455X (gfx1250) — compile-verified, matrix operand order checked by reading, not run on hardware
//
#include <hip/hip_runtime.h>
#include <hip/hip_bf16.h>

// ---------------------------------------------------------------------------
// HiLo windowed transformer encoder for MI455X (gfx1250, wave32, WMMA).
// All GEMMs + both attention einsums on v_wmma_f32_16x16x32_bf16, with
// fragment operands fed by contiguous b128 loads (weights pre-swizzled into
// WMMA fragment order; activations fragment-gathered as 2x16B runs per lane).
// ---------------------------------------------------------------------------

typedef __bf16 bf16_t;
typedef __attribute__((ext_vector_type(16))) __bf16 v16bf;
typedef __attribute__((ext_vector_type(8)))  __bf16 v8bf;
typedef __attribute__((ext_vector_type(8)))  float  v8f;

#define DEVI __device__ __forceinline__

DEVI v8f wmma_bf16(v16bf a, v16bf b, v8f c) {
  // (neg_a, A, neg_b, B, c_mod, C, reuse_a, reuse_b)
  return __builtin_amdgcn_wmma_f32_16x16x32_bf16(false, a, false, b, (short)0, c, false, false);
}

DEVI v8f vzero8() {
  v8f z;
#pragma unroll
  for (int i = 0; i < 8; ++i) z[i] = 0.f;
  return z;
}

DEVI v16bf make_frag(v8bf lo, v8bf hi) {
  return __builtin_shufflevector(lo, hi, 0, 1, 2, 3, 4, 5, 6, 7,
                                 8, 9, 10, 11, 12, 13, 14, 15);
}

// A fragment (16x32 bf16 MxK) direct from a row-major [.. x ld] matrix.
// ISA layout: lane&15 = row; lane half selects K 0-7/16-23 vs 8-15/24-31.
// Both 8-element runs are 16B-aligned when ld%8==0 and base is 16B-aligned.
DEVI v16bf load_a_rowmajor(const bf16_t* __restrict__ base, int row0, int k0, int ld) {
  int lane = threadIdx.x & 31;
  const bf16_t* p =
      base + (size_t)(row0 + (lane & 15)) * ld + k0 + ((lane & 16) ? 8 : 0);
  v8bf lo = *(const v8bf*)p;
  v8bf hi = *(const v8bf*)(p + 16);
  return make_frag(lo, hi);
}

// Fragment from swizzled storage: 32 lanes x 16 contiguous bf16 (1KB/fragment).
DEVI v16bf load_frag_swz(const bf16_t* __restrict__ frag) {
  int lane = threadIdx.x & 31;
  const v8bf* p = (const v8bf*)(frag + lane * 16);
  return make_frag(p[0], p[1]);
}

// ---------------------------------------------------------------------------
// GEMM: C[M x N] = act(A[M x K] @ W + bias); W pre-swizzled B-fragment order
// [K/32][N/16][lane][16]. Block 256 thr = 8 waves (4x2), wave tile 32x64.
// No LDS, no barriers: A 2x b128/frag from global, B 2x b128/frag from global.
// ---------------------------------------------------------------------------
template <bool HAS_BIAS, bool RELU, bool OUT_BF16>
__global__ __launch_bounds__(256) void gemm_bf16_kernel(
    const bf16_t* __restrict__ A, const bf16_t* __restrict__ Wz,
    const float* __restrict__ bias, void* __restrict__ Cout,
    int ldc, int coff, int M, int N, int K) {
  const int tid = threadIdx.x;
  const int wid = tid >> 5, lane = tid & 31;
  const int wrow = wid & 3, wcol = wid >> 2;
  const int bm = blockIdx.x * 128 + wrow * 32;
  const int bn0 = blockIdx.y * 128 + wcol * 64;
  const int nfrag = N >> 4;

  v8f acc[2][4];
#pragma unroll
  for (int i = 0; i < 2; ++i)
#pragma unroll
    for (int j = 0; j < 4; ++j) acc[i][j] = vzero8();

  for (int k0 = 0; k0 < K; k0 += 32) {
    v16bf a0 = load_a_rowmajor(A, bm, k0, K);
    v16bf a1 = load_a_rowmajor(A, bm + 16, k0, K);
    const bf16_t* wz = Wz + ((size_t)(k0 >> 5) * nfrag + (bn0 >> 4)) * 512;
#pragma unroll
    for (int nt = 0; nt < 4; ++nt) {
      v16bf bfr = load_frag_swz(wz + nt * 512);
      acc[0][nt] = wmma_bf16(a0, bfr, acc[0][nt]);
      acc[1][nt] = wmma_bf16(a1, bfr, acc[1][nt]);
    }
  }

  const int hh = (lane >> 4) & 1;
  const int cc = lane & 15;
#pragma unroll
  for (int mt = 0; mt < 2; ++mt) {
#pragma unroll
    for (int nt = 0; nt < 4; ++nt) {
      int colN = bn0 + nt * 16 + cc;
      float bi = HAS_BIAS ? bias[colN] : 0.f;
      int rowBase = bm + mt * 16 + hh * 8;
#pragma unroll
      for (int v = 0; v < 8; ++v) {
        float val = acc[mt][nt][v] + bi;
        if (RELU) val = fmaxf(val, 0.f);
        size_t off = (size_t)(rowBase + v) * ldc + coff + colN;
        if (OUT_BF16)
          ((bf16_t*)Cout)[off] = (bf16_t)val;
        else
          ((float*)Cout)[off] = val;
      }
    }
  }
}

// ---------------------------------------------------------------------------
// lofi attention per (window, head): Q[N x 32] vs pooled K/V[M x 32].
// Head dim 32 == WMMA K: one WMMA per 16x16 S tile. Q/K fragments direct from
// global (contiguous per-lane runs). V staged once in LDS pre-swizzled to
// B-fragment order; P written per-wave in A-fragment-swizzled order.
// ---------------------------------------------------------------------------
template <int MT>  // MT = M/16 (8 for layer 0, 2 for layer 1)
__global__ __launch_bounds__(256) void lofi_attn_kernel(
    const bf16_t* __restrict__ Q, const bf16_t* __restrict__ KV,
    bf16_t* __restrict__ O, int N, int Mp, float scale) {
  constexpr int M = MT * 16;
  constexpr int KT = MT / 2;  // K-steps of 32 over M
  __shared__ __align__(16) bf16_t Vs[KT * 2 * 512];
  __shared__ __align__(16) bf16_t Ps[8][KT * 512];

  const int win = blockIdx.x, head = blockIdx.y;
  const int tid = threadIdx.x, wid = tid >> 5, lane = tid & 31;
  const int hh = (lane >> 4) & 1, cc = lane & 15;

  // Stage V (M x 32) into B-fragment swizzle: [kt][nt][lane][16]
  for (int i = tid; i < M * 32; i += 256) {
    int m = i >> 5, d = i & 31;
    int lane2 = (d & 15) | (((m >> 3) & 1) << 4);
    int idx = (m & 7) | ((m & 16) >> 1);
    Vs[(((m >> 5) * 2 + (d >> 4)) * 32 + lane2) * 16 + idx] =
        KV[(size_t)(win * Mp + m) * 256 + 128 + head * 32 + d];
  }
  __syncthreads();

  // K^T B-fragments == A-style gathers of K rows: preload once.
  v16bf kf[MT];
#pragma unroll
  for (int mt = 0; mt < MT; ++mt)
    kf[mt] = load_a_rowmajor(KV + head * 32, win * Mp + mt * 16, 0, 256);

  for (int s0 = 0; s0 < N; s0 += 128) {
    v16bf qf = load_a_rowmajor(Q + head * 32, win * N + s0 + wid * 16, 0, 128);
    v8f sf[MT];
#pragma unroll
    for (int mt = 0; mt < MT; ++mt) sf[mt] = wmma_bf16(qf, kf[mt], vzero8());

    // row softmax over M keys (row = hh*8+v; key col = mt*16+cc)
    float rmax[8], rsum[8];
#pragma unroll
    for (int v = 0; v < 8; ++v) {
      float mx = -1e30f;
#pragma unroll
      for (int mt = 0; mt < MT; ++mt) {
        sf[mt][v] *= scale;
        mx = fmaxf(mx, sf[mt][v]);
      }
#pragma unroll
      for (int o = 8; o >= 1; o >>= 1) mx = fmaxf(mx, __shfl_xor(mx, o, 16));
      rmax[v] = mx;
    }
#pragma unroll
    for (int v = 0; v < 8; ++v) {
      float s = 0.f;
#pragma unroll
      for (int mt = 0; mt < MT; ++mt) {
        float e = __expf(sf[mt][v] - rmax[v]);
        sf[mt][v] = e;
        s += e;
      }
#pragma unroll
      for (int o = 8; o >= 1; o >>= 1) s += __shfl_xor(s, o, 16);
      rsum[v] = 1.f / s;
    }

    // Store P in A-fragment swizzle: element (row, k=mt*16+cc)
    bf16_t* pw = &Ps[wid][0];
#pragma unroll
    for (int mt = 0; mt < MT; ++mt)
#pragma unroll
      for (int v = 0; v < 8; ++v) {
        int lane2 = (hh * 8 + v) | (((cc >> 3) & 1) << 4);
        int idx = (cc & 7) | ((mt & 1) << 3);
        pw[((mt >> 1) * 32 + lane2) * 16 + idx] = (bf16_t)(sf[mt][v] * rsum[v]);
      }

    // O[16 x 32] = P[16 x M] @ V[M x 32]
    v8f oacc[2] = {vzero8(), vzero8()};
#pragma unroll
    for (int ks = 0; ks < KT; ++ks) {
      v16bf pa = load_frag_swz(pw + ks * 512);
#pragma unroll
      for (int j = 0; j < 2; ++j) {
        v16bf vb = load_frag_swz(&Vs[(ks * 2 + j) * 512]);
        oacc[j] = wmma_bf16(pa, vb, oacc[j]);
      }
    }
#pragma unroll
    for (int j = 0; j < 2; ++j)
#pragma unroll
      for (int v = 0; v < 8; ++v) {
        int row = s0 + wid * 16 + hh * 8 + v;
        O[(size_t)(win * N + row) * 128 + head * 32 + j * 16 + cc] =
            (bf16_t)oacc[j][v];
      }
  }
}

// ---------------------------------------------------------------------------
// Elementwise / data-movement kernels
// ---------------------------------------------------------------------------

// win_partition + positional add: canonical [B,C,128,128] -> window-major
// tokens, fp32 residual + bf16 GEMM operand.
__global__ void gather_kernel(const float* __restrict__ X, const float* __restrict__ P,
                              float* __restrict__ R, bf16_t* __restrict__ Xb,
                              int wh, int ww, int nWb) {
  size_t idx = (size_t)blockIdx.x * 256 + threadIdx.x;
  int w = idx & 127, h = (int)(idx >> 7) & 127, c = (int)(idx >> 14) & 255;
  int b = (int)(idx >> 22);
  float v = X[idx] + P[idx];
  int nHb = 128 / wh;
  int win = (b * nHb + h / wh) * nWb + w / ww;
  int n = (h % wh) * ww + (w % ww);
  size_t t = (size_t)win * (wh * ww) + n;
  R[t * 256 + c] = v;
  Xb[t * 256 + c] = (bf16_t)v;
}

// 2x2 token mean (lofi key/value pooling), fp32 -> bf16
__global__ void pool_kernel(const float* __restrict__ R, bf16_t* __restrict__ Xs,
                            int N, int ww, int Mp) {
  int ts = blockIdx.x, c = threadIdx.x;
  int win = ts / Mp, m = ts % Mp;
  int wwp = ww >> 1;
  size_t base = (size_t)win * N + (size_t)((m / wwp) * 2) * ww + (m % wwp) * 2;
  float v = 0.25f * (R[base * 256 + c] + R[(base + 1) * 256 + c] +
                     R[(base + ww) * 256 + c] + R[(base + ww + 1) * 256 + c]);
  Xs[(size_t)ts * 256 + c] = (bf16_t)v;
}

// hifi: avgpool3 over a 2x2 block (count_include_pad=False) == block mean,
// minus identity.
__global__ void hifi_diff_kernel(const bf16_t* __restrict__ Hh, bf16_t* __restrict__ Hw,
                                 int N, int ww) {
  int g = blockIdx.x, d = threadIdx.x;  // 128 threads = HD channels
  int gpw = N >> 2;
  int win = g / gpw, m = g % gpw;
  int wwp = ww >> 1;
  size_t base = (size_t)win * N + (size_t)((m / wwp) * 2) * ww + (m % wwp) * 2;
  size_t t0 = base, t1 = base + 1, t2 = base + ww, t3 = base + ww + 1;
  float a = (float)Hh[t0 * 128 + d], b = (float)Hh[t1 * 128 + d];
  float c = (float)Hh[t2 * 128 + d], e = (float)Hh[t3 * 128 + d];
  float mean = 0.25f * (a + b + c + e);
  Hw[t0 * 128 + d] = (bf16_t)(mean - a);
  Hw[t1 * 128 + d] = (bf16_t)(mean - b);
  Hw[t2 * 128 + d] = (bf16_t)(mean - c);
  Hw[t3 * 128 + d] = (bf16_t)(mean - e);
}

// LayerNorm(X + Y), wave-per-token; optional second LN; optional fp32/bf16
// token-major outputs; optional win_reverse scatter to canonical layout.
__global__ __launch_bounds__(256) void ln_kernel(
    const float* __restrict__ X, const float* __restrict__ Y,
    const float* __restrict__ g1, const float* __restrict__ b1,
    const float* __restrict__ g2, const float* __restrict__ b2,
    float* __restrict__ outF, bf16_t* __restrict__ outB,
    float* __restrict__ outCanon, int wh, int ww, int nWb) {
  int t = blockIdx.x * 8 + (threadIdx.x >> 5);
  int lane = threadIdx.x & 31;
  size_t base = (size_t)t * 256;
  float v[8];
  float s = 0.f;
#pragma unroll
  for (int i = 0; i < 8; ++i) {
    int c = lane + i * 32;
    v[i] = X[base + c] + Y[base + c];
    s += v[i];
  }
#pragma unroll
  for (int o = 16; o >= 1; o >>= 1) s += __shfl_xor(s, o, 32);
  float mean = s * (1.f / 256.f), q = 0.f;
#pragma unroll
  for (int i = 0; i < 8; ++i) { float d = v[i] - mean; q += d * d; }
#pragma unroll
  for (int o = 16; o >= 1; o >>= 1) q += __shfl_xor(q, o, 32);
  float inv = rsqrtf(q * (1.f / 256.f) + 1e-5f);
#pragma unroll
  for (int i = 0; i < 8; ++i) {
    int c = lane + i * 32;
    v[i] = (v[i] - mean) * inv * g1[c] + b1[c];
  }
  if (g2) {  // final LayerNorm (last layer only)
    float s2 = 0.f;
#pragma unroll
    for (int i = 0; i < 8; ++i) s2 += v[i];
#pragma unroll
    for (int o = 16; o >= 1; o >>= 1) s2 += __shfl_xor(s2, o, 32);
    float m2 = s2 * (1.f / 256.f), q2 = 0.f;
#pragma unroll
    for (int i = 0; i < 8; ++i) { float d = v[i] - m2; q2 += d * d; }
#pragma unroll
    for (int o = 16; o >= 1; o >>= 1) q2 += __shfl_xor(q2, o, 32);
    float inv2 = rsqrtf(q2 * (1.f / 256.f) + 1e-5f);
#pragma unroll
    for (int i = 0; i < 8; ++i) {
      int c = lane + i * 32;
      v[i] = (v[i] - m2) * inv2 * g2[c] + b2[c];
    }
  }
#pragma unroll
  for (int i = 0; i < 8; ++i) {
    int c = lane + i * 32;
    if (outF) outF[base + c] = v[i];
    if (outB) outB[base + c] = (bf16_t)v[i];
  }
  if (outCanon) {
    int Nw = wh * ww;
    int win = t / Nw, n = t % Nw;
    int nHb = 128 / wh;
    int b = win / (nHb * nWb), rem = win % (nHb * nWb);
    int h = (rem / nWb) * wh + n / ww;
    int w = (rem % nWb) * ww + n % ww;
#pragma unroll
    for (int i = 0; i < 8; ++i) {
      int c = lane + i * 32;
      outCanon[(((size_t)b * 256 + c) * 128 + h) * 128 + w] = v[i];
    }
  }
}

// f32 weight [K x N] -> bf16 in WMMA B-fragment swizzle [K/32][N/16][lane][16]
__global__ void cvt_swzB_kernel(const float* __restrict__ src,
                                bf16_t* __restrict__ dst, int K, int N) {
  int i = blockIdx.x * 256 + threadIdx.x;
  if (i >= K * N) return;
  int k = i / N, c = i % N;
  int lane = (c & 15) | (((k >> 3) & 1) << 4);
  int idx = (k & 7) | ((k & 16) >> 1);
  dst[(((size_t)(k >> 5) * (N >> 4) + (c >> 4)) * 32 + lane) * 16 + idx] =
      (bf16_t)src[i];
}

// ---------------------------------------------------------------------------
// Host side
// ---------------------------------------------------------------------------

static void launch_gemm(hipStream_t st, const bf16_t* A, const bf16_t* Wz,
                        const float* bias, void* C, int ldc, int coff, int M,
                        int N, int K, bool relu, bool outbf16) {
  dim3 g(M / 128, N / 128);
  if (bias && relu && outbf16)
    gemm_bf16_kernel<true, true, true><<<g, 256, 0, st>>>(A, Wz, bias, C, ldc, coff, M, N, K);
  else if (bias && !relu && !outbf16)
    gemm_bf16_kernel<true, false, false><<<g, 256, 0, st>>>(A, Wz, bias, C, ldc, coff, M, N, K);
  else
    gemm_bf16_kernel<false, false, true><<<g, 256, 0, st>>>(A, Wz, bias, C, ldc, coff, M, N, K);
}

// Per-layer param indices, assuming JAX pytree (sorted-key) flatten order:
// layers[i]: b1 b2 h_proj_b h_proj_w h_qkv l_kv l_proj_b l_proj_w l_q
//            n1_b n1_g n2_b n2_g w1 w2 ; then nf_b, nf_g.
enum { I_B1 = 0, I_B2, I_HPROJ_B, I_HPROJ_W, I_HQKV, I_LKV, I_LPROJ_B,
       I_LPROJ_W, I_LQ, I_N1B, I_N1G, I_N2B, I_N2G, I_W1, I_W2 };

extern "C" void kernel_launch(void* const* d_in, const int* in_sizes, int n_in,
                              void* d_out, int out_size, void* d_ws, size_t ws_size,
                              hipStream_t stream) {
  (void)in_sizes; (void)n_in; (void)out_size; (void)ws_size;
  const int T = 8 * 128 * 128;  // tokens
  const int Tp = T / 4;

  char* ws = (char*)d_ws;
  // workspace layout (bytes), ~505 MB with buffer reuse
  float*  R    = (float*)(ws + 0);                       // fp32 residual [T x 256]
  float*  BIGf = (float*)(ws + 134217728);               // Aout f32 [T x 256]
  bf16_t* BIGb = (bf16_t*)(ws + 134217728);              // F1 bf16 [T x 512] (aliases)
  bf16_t* Xb   = (bf16_t*)(ws + 268435456);              // bf16 tokens [T x 256]
  bf16_t* Qb   = (bf16_t*)(ws + 335544320);              // [T x 128]
  bf16_t* Hh   = (bf16_t*)(ws + 335544320 + 33554432);   // [T x 128]
  bf16_t* Hw   = (bf16_t*)(ws + 335544320 + 67108864);   // [T x 128]
  bf16_t* Ol   = (bf16_t*)(ws + 335544320 + 100663296);  // [T x 128]
  float*  F2   = (float*)(ws + 335544320);               // f32 [T x 256] (aliases Q/Hh/Hw/Ol)
  bf16_t* Xs   = (bf16_t*)(ws + 469762048);              // [Tp x 256]
  bf16_t* KVb  = (bf16_t*)(ws + 469762048 + 16777216);   // [Tp x 256]
  bf16_t* wb   = (bf16_t*)(ws + 503316480);              // swizzled bf16 weights
  bf16_t* Wlq = wb, *Wlkv = wb + 32768, *Wlproj = wb + 98304;
  bf16_t* Whqkv = wb + 114688, *Whproj = wb + 147456;
  bf16_t* Ww1 = wb + 163840, *Ww2 = wb + 294912;

  const float* src = (const float*)d_in[0];
  const float* pos = (const float*)d_in[1];
  float* outC = (float*)d_out;

  const int WH[2] = {16, 8}, WW[2] = {32, 16};
  const float SCALE = 0.17677669529663687f;  // 32^-0.5

  for (int L = 0; L < 2; ++L) {
    const int wh = WH[L], ww = WW[L];
    const int N = wh * ww, Mp = N / 4;
    const int nWb = 128 / ww;
    const int NW = T / N;
    const int base = 2 + L * 15;
    const float* p[15];
    for (int i = 0; i < 15; ++i) p[i] = (const float*)d_in[base + i];
    const float* nf_b = (const float*)d_in[2 + 2 * 15 + 0];
    const float* nf_g = (const float*)d_in[2 + 2 * 15 + 1];

    // weights -> bf16, swizzled to B-fragment order
    cvt_swzB_kernel<<<128, 256, 0, stream>>>(p[I_LQ], Wlq, 256, 128);
    cvt_swzB_kernel<<<256, 256, 0, stream>>>(p[I_LKV], Wlkv, 256, 256);
    cvt_swzB_kernel<<<64, 256, 0, stream>>>(p[I_LPROJ_W], Wlproj, 128, 128);
    cvt_swzB_kernel<<<128, 256, 0, stream>>>(p[I_HQKV], Whqkv, 256, 128);
    cvt_swzB_kernel<<<64, 256, 0, stream>>>(p[I_HPROJ_W], Whproj, 128, 128);
    cvt_swzB_kernel<<<512, 256, 0, stream>>>(p[I_W1], Ww1, 256, 512);
    cvt_swzB_kernel<<<512, 256, 0, stream>>>(p[I_W2], Ww2, 512, 256);

    // window-partition + pos add
    const float* canon = (L == 0) ? src : outC;
    gather_kernel<<<(T * 256) / 256, 256, 0, stream>>>(canon, pos, R, Xb, wh, ww, nWb);

    // lofi Q, hifi h, pooled keys, lofi KV
    launch_gemm(stream, Xb, Wlq, nullptr, Qb, 128, 0, T, 128, 256, false, true);
    launch_gemm(stream, Xb, Whqkv, nullptr, Hh, 128, 0, T, 128, 256, false, true);
    pool_kernel<<<Tp, 256, 0, stream>>>(R, Xs, N, ww, Mp);
    launch_gemm(stream, Xs, Wlkv, nullptr, KVb, 256, 0, Tp, 256, 256, false, true);

    // hifi branch: (block mean - x) then projection -> Aout[:, 0:128)
    hifi_diff_kernel<<<T / 4, 128, 0, stream>>>(Hh, Hw, N, ww);
    launch_gemm(stream, Hw, Whproj, p[I_HPROJ_B], BIGf, 256, 0, T, 128, 128, false, false);

    // lofi attention + projection -> Aout[:, 128:256)
    if (Mp == 128)
      lofi_attn_kernel<8><<<dim3(NW, 4), 256, 0, stream>>>(Qb, KVb, Ol, N, Mp, SCALE);
    else
      lofi_attn_kernel<2><<<dim3(NW, 4), 256, 0, stream>>>(Qb, KVb, Ol, N, Mp, SCALE);
    launch_gemm(stream, Ol, Wlproj, p[I_LPROJ_B], BIGf, 256, 128, T, 128, 128, false, false);

    // LN1: s1 = LN(R + Aout); fp32 in-place over R, bf16 into Xb
    ln_kernel<<<T / 8, 256, 0, stream>>>(R, BIGf, p[I_N1G], p[I_N1B], nullptr, nullptr,
                                         R, Xb, nullptr, wh, ww, nWb);

    // FFN: F1 = relu(s1 @ w1 + b1) bf16; F2 = F1 @ w2 + b2 f32
    launch_gemm(stream, Xb, Ww1, p[I_B1], BIGb, 512, 0, T, 512, 256, true, true);
    launch_gemm(stream, BIGb, Ww2, p[I_B2], F2, 256, 0, T, 256, 512, false, false);

    // LN2 (+ final LN on last layer) + win_reverse scatter to canonical out
    ln_kernel<<<T / 8, 256, 0, stream>>>(R, F2, p[I_N2G], p[I_N2B],
                                         (L == 1) ? nf_g : nullptr,
                                         (L == 1) ? nf_b : nullptr,
                                         nullptr, nullptr, outC, wh, ww, nWb);
  }
}
